// PerformerSeperator_9861244911653
// MI455X (gfx1250) — compile-verified, matrix operand order checked by reading, not run on hardware
//
#include <hip/hip_runtime.h>
#include <cmath>

typedef __bf16 bf16;
typedef __attribute__((ext_vector_type(16))) bf16  v16bf;
typedef __attribute__((ext_vector_type(8)))  bf16  v8bf;
typedef __attribute__((ext_vector_type(8)))  float v8f;

#define DDIM  768
#define NTOK  512
#define BATCH 8
#define HEADS 8
#define DH    96
#define MFEAT 256
#define FFDIM 3072
#define ROWS  (BATCH*NTOK)   // 4096
#define QKVW  (3*DDIM)       // 2304

// ---------------- helpers ----------------
__device__ inline void atomicMaxF(float* addr, float v) {
  if (v >= 0.f) atomicMax((int*)addr, __float_as_int(v));
  else          atomicMin((unsigned int*)addr, __float_as_uint(v));
}

template<typename TB> __device__ inline v8bf ldcvt8(const TB* p);
template<> __device__ inline v8bf ldcvt8<bf16>(const bf16* p) { return *(const v8bf*)p; }
template<> __device__ inline v8bf ldcvt8<float>(const float* p) {
  float4 f0 = ((const float4*)p)[0];
  float4 f1 = ((const float4*)p)[1];
  v8bf r;
  r[0] = (bf16)f0.x; r[1] = (bf16)f0.y; r[2] = (bf16)f0.z; r[3] = (bf16)f0.w;
  r[4] = (bf16)f1.x; r[5] = (bf16)f1.y; r[6] = (bf16)f1.z; r[7] = (bf16)f1.w;
  return r;
}

// CDNA5 async global->LDS copy (32B per lane = two b128 sharing one address pair;
// IOFFSET applies to both the global and the LDS address). Tracked by ASYNCcnt.
__device__ inline void async_ld_b128x2(unsigned lds_addr, const void* gaddr) {
  asm volatile("global_load_async_to_lds_b128 %0, %1, off\n\t"
               "global_load_async_to_lds_b128 %0, %1, off offset:16"
               :: "v"(lds_addr), "v"(gaddr) : "memory");
}

// ---------------- WMMA GEMM ----------------
// C[z][m][n] = sum_k A[z][m][k] * B[z][k][n]
// A bf16 (M,K) row-major, M % 128 == 0 (no row guards anywhere).
// BT:  B stored (N,K) row-major bf16, N % 128 == 0 (no guards; async-staged).
// !BT: B stored (K,N) row-major, N % 16 == 0 (whole-chunk guard; sync transpose staging).
// LDS layout for both operands: (rows x 32K) bf16, row stride 40 halves (80B, 16B aligned).
#define BM 128
#define BN 128
#define BK 32
#define LSTR 40

__device__ inline v16bf frag_ld(const bf16* base, int row0, int lane) {
  int r = row0 + (lane & 15);
  const bf16* p = base + r * LSTR + ((lane >> 4) << 3);  // halves {0..7,16..23} / {8..15,24..31}
  v8bf lo = *(const v8bf*)(p);
  v8bf hi = *(const v8bf*)(p + 16);
  return __builtin_shufflevector(lo, hi, 0,1,2,3,4,5,6,7,8,9,10,11,12,13,14,15);
}

// EPI: 0 = store f32 (guarded c<N), 1 = bf16(acc+bias), 2 = f32 C += acc+bias, 3 = bf16 gelu(acc+bias)
template<typename TB, bool BT, int EPI, bool MAXT>
__global__ __launch_bounds__(256) void gemm_k(
    const bf16* __restrict__ A, long aSB, long aSH, int lda,
    const TB*  __restrict__ B, long bSB, long bSH, int ldb,
    void* __restrict__ C, long cSB, long cSH, int ldc,
    const float* __restrict__ bias,
    int M, int N, int K, int H2, float* gmaxp)
{
  __shared__ alignas(16) bf16 As[2][BM * LSTR];
  __shared__ alignas(16) bf16 Bs[2][BN * LSTR];
  const int tid  = threadIdx.x;
  const int lane = tid & 31, wave = tid >> 5;
  const int wm = (wave & 1) * 64, wn = (wave >> 1) * 32;
  const int z  = blockIdx.z;
  const int zb = z / H2, zh = z % H2;
  const long aOff = (long)zb * aSB + (long)zh * aSH;
  const long bOff = (long)zb * bSB + (long)zh * bSH;
  const long cOff = (long)zb * cSB + (long)zh * cSH;
  const int bm = blockIdx.y * BM, bn = blockIdx.x * BN;

  v8f acc[4][2] = {};

  // staging coordinates (per thread: one 16-element row chunk of A and of B)
  const int ar = tid >> 1, ac = (tid & 1) << 4;
  const bf16* Ag = A + aOff + (long)(bm + ar) * lda + ac;
  const unsigned aL[2] = { (unsigned)(size_t)&As[0][ar * LSTR + ac],
                           (unsigned)(size_t)&As[1][ar * LSTR + ac] };
  const TB* BgT = B + bOff + (long)(bn + ar) * ldb + ac;   // BT mode: (N,K) rows (bf16 only)
  const unsigned bL[2] = { (unsigned)(size_t)&Bs[0][ar * LSTR + ac],
                           (unsigned)(size_t)&Bs[1][ar * LSTR + ac] };
  const int kr = tid >> 3, nc = (tid & 7) << 4;            // !BT mode: 32 K-rows x 128 N
  const TB* BgN = B + bOff + (long)kr * ldb + bn + nc;
  const bool nok = (bn + nc + 16) <= N;

  v8bf b0 = {}, b1 = {};
  // prologue: tile 0
  async_ld_b128x2(aL[0], Ag);
  if (BT) {
    async_ld_b128x2(bL[0], BgT);
  } else {
    if (nok) { b0 = ldcvt8<TB>(BgN); b1 = ldcvt8<TB>(BgN + 8); }
  }

  const int ntiles = K / BK;
  for (int t = 0; t < ntiles; ++t) {
    const int cur = t & 1, nxt = cur ^ 1;
    if (!BT) {   // transpose scatter of B tile t into LDS
      #pragma unroll
      for (int j = 0; j < 8; ++j) Bs[cur][(nc + j) * LSTR + kr] = b0[j];
      #pragma unroll
      for (int j = 0; j < 8; ++j) Bs[cur][(nc + 8 + j) * LSTR + kr] = b1[j];
    }
    const bool more = (t + 1) < ntiles;
    if (more) {  // issue async loads for tile t+1 into the other buffer
      const int kn = (t + 1) * BK;
      async_ld_b128x2(aL[nxt], Ag + kn);
      if (BT) async_ld_b128x2(bL[nxt], BgT + kn);
    }
    // wait for tile t's asyncs (tile t+1's stay in flight), then join all waves
    if (more) {
      if (BT) asm volatile("s_wait_asynccnt 4" ::: "memory");
      else    asm volatile("s_wait_asynccnt 2" ::: "memory");
    } else {
      asm volatile("s_wait_asynccnt 0" ::: "memory");
    }
    __syncthreads();

    if (!BT && more) {  // register-prefetch next B tile (hides under WMMA chain)
      const TB* bp = BgN + (long)((t + 1) * BK) * ldb;
      if (nok) { b0 = ldcvt8<TB>(bp); b1 = ldcvt8<TB>(bp + 8); }
      else     { b0 = {};  b1 = {}; }
    }

    v16bf af[4], bfm[2];
    #pragma unroll
    for (int i = 0; i < 4; ++i) af[i]  = frag_ld(As[cur], wm + i * 16, lane);
    #pragma unroll
    for (int j = 0; j < 2; ++j) bfm[j] = frag_ld(Bs[cur], wn + j * 16, lane);
    #pragma unroll
    for (int i = 0; i < 4; ++i)
      #pragma unroll
      for (int j = 0; j < 2; ++j)
        acc[i][j] = __builtin_amdgcn_wmma_f32_16x16x32_bf16(
            false, af[i], false, bfm[j], (short)0, acc[i][j], false, false);
    __syncthreads();
  }

  // epilogue: C layout = lanes 0-15: M=v,N=lane; lanes 16-31: M=8+v,N=lane-16
  const int lane15 = lane & 15;
  const int rowAdd = (lane >> 4) << 3;
  float mx = -3.0e38f;
  #pragma unroll
  for (int i = 0; i < 4; ++i) {
    const int r0 = bm + wm + i * 16 + rowAdd;
    #pragma unroll
    for (int j = 0; j < 2; ++j) {
      const int c = bn + wn + j * 16 + lane15;
      const float bv = (EPI != 0) ? bias[c] : 0.f;   // biased epilogues always have N%128==0
      #pragma unroll
      for (int v = 0; v < 8; ++v) {
        const int r = r0 + v;
        const float val = acc[i][j][v];
        const long off = cOff + (long)r * ldc + c;
        if (EPI == 0)      { if (c < N) ((float*)C)[off] = val; }
        else if (EPI == 1) ((bf16*)C)[off]  = (bf16)(val + bv);
        else if (EPI == 2) ((float*)C)[off] += val + bv;
        else {
          const float zz = val + bv;
          ((bf16*)C)[off] = (bf16)(0.5f * zz * (1.f + erff(zz * 0.70710678f)));
        }
        if (MAXT) mx = fmaxf(mx, val);
      }
    }
  }
  if (MAXT) {
    #pragma unroll
    for (int o = 16; o > 0; o >>= 1) mx = fmaxf(mx, __shfl_xor(mx, o, 32));
    if (lane == 0) atomicMaxF(gmaxp, mx);
  }
}

// ---------------- scalar kernels ----------------
__global__ __launch_bounds__(256) void patch_k(const float* __restrict__ mel,
    const float* __restrict__ pw, const float* __restrict__ pb,
    const float* __restrict__ pos, float* __restrict__ x) {
  int idx = blockIdx.x * 256 + threadIdx.x;
  if (idx >= BATCH * NTOK * DDIM) return;
  int d = idx % DDIM, n = (idx / DDIM) % NTOK, b = idx / (DDIM * NTOK);
  int ph = n >> 3, pwc = n & 7;
  const float* mp = mel + ((long)b * 1024 + ph * 16) * 128 + pwc * 16;
  const float* wp = pw + (long)d * 256;
  float acc = 0.f;
  #pragma unroll 4
  for (int i = 0; i < 16; ++i)
    #pragma unroll
    for (int j = 0; j < 16; ++j)
      acc += mp[i * 128 + j] * wp[i * 16 + j];
  x[idx] = acc + pb[d] + pos[(long)n * DDIM + d];
}

__global__ __launch_bounds__(256) void ln_k(const float* __restrict__ x,
    const float* __restrict__ s, const float* __restrict__ bsh, bf16* __restrict__ out) {
  int row = blockIdx.x, t = threadIdx.x;
  const float* xr = x + (long)row * DDIM;
  float v0 = xr[t], v1 = xr[t + 256], v2 = xr[t + 512];
  __shared__ float r1[256], r2[256];
  r1[t] = v0 + v1 + v2;
  r2[t] = v0 * v0 + v1 * v1 + v2 * v2;
  __syncthreads();
  for (int o = 128; o > 0; o >>= 1) {
    if (t < o) { r1[t] += r1[t + o]; r2[t] += r2[t + o]; }
    __syncthreads();
  }
  float mu = r1[0] * (1.f / 768.f);
  float var = r2[0] * (1.f / 768.f) - mu * mu;
  float rs = rsqrtf(var + 1e-5f);
  bf16* orow = out + (long)row * DDIM;
  orow[t]       = (bf16)((v0 - mu) * rs * s[t]       + bsh[t]);
  orow[t + 256] = (bf16)((v1 - mu) * rs * s[t + 256] + bsh[t + 256]);
  orow[t + 512] = (bf16)((v2 - mu) * rs * s[t + 512] + bsh[t + 512]);
}

// per-(b,h,n): pq row from u_q (row-max stabilizer), also diag_k for the k pass
__global__ __launch_bounds__(256) void phi_q_k(const float* __restrict__ u,
    const bf16* __restrict__ qkv, bf16* __restrict__ pq, float* __restrict__ diagk) {
  int idx = blockIdx.x * 256 + threadIdx.x;
  if (idx >= BATCH * HEADS * NTOK) return;
  int z = idx >> 9, n = idx & 511;
  int b = z >> 3, h = z & 7;
  const float* ur = u + ((long)z * NTOK + n) * MFEAT;
  float mx = -3.0e38f;
  #pragma unroll 8
  for (int m = 0; m < MFEAT; ++m) mx = fmaxf(mx, ur[m]);
  const bf16* qr = qkv + ((long)b * NTOK + n) * QKVW + h * DH;
  float sq = 0.f;
  for (int e = 0; e < DH; ++e) { float q = (float)qr[e]; sq += q * q; }
  const float inv_sqrt_d = 0.10206207f;            // 96^-0.5 (xd = x * 96^-0.25)
  const float diag = 0.5f * sq * inv_sqrt_d;
  bf16* pr = pq + ((long)z * NTOK + n) * MFEAT;
  const float sm = 0.0625f;                        // 256^-0.5
  for (int m = 0; m < MFEAT; ++m)
    pr[m] = (bf16)((expf(ur[m] - diag - mx) + 1e-4f) * sm);
  const bf16* kr = qr + DDIM;                      // k slice, same row
  float sk = 0.f;
  for (int e = 0; e < DH; ++e) { float kv = (float)kr[e]; sk += kv * kv; }
  diagk[idx] = 0.5f * sk * inv_sqrt_d;
}

__global__ void init_gmax(float* g) { *g = -3.0e38f; }

// per-(b,h,m): pk column -> pkT row (bf16) + ksum
__global__ __launch_bounds__(256) void phi_k_k(const float* __restrict__ u,
    const float* __restrict__ diagk, const float* __restrict__ gmaxp,
    bf16* __restrict__ pkT, float* __restrict__ ksum) {
  int idx = blockIdx.x * 256 + threadIdx.x;
  if (idx >= BATCH * HEADS * MFEAT) return;
  int z = idx >> 8, m = idx & 255;
  const float g = *gmaxp;
  const float* uc = u + (long)z * NTOK * MFEAT + m;
  const float* dg = diagk + (long)z * NTOK;
  bf16* pr = pkT + ((long)z * MFEAT + m) * NTOK;
  const float sm = 0.0625f;
  float s = 0.f;
  for (int n = 0; n < NTOK; ++n) {
    float v = (expf(uc[(long)n * MFEAT] - dg[n] - g) + 1e-4f) * sm;
    pr[n] = (bf16)v;
    s += v;
  }
  ksum[idx] = s;
}

// per-(b,h,n): den = pq . ksum ; attn[b,n,h*96+e] = out/den  (bf16, head-merged)
__global__ __launch_bounds__(256) void merge_k(const float* __restrict__ obuf,
    const bf16* __restrict__ pq, const float* __restrict__ ksum, bf16* __restrict__ attn) {
  int idx = blockIdx.x * 256 + threadIdx.x;
  if (idx >= BATCH * HEADS * NTOK) return;
  int z = idx >> 9, n = idx & 511;
  int b = z >> 3, h = z & 7;
  const bf16* pr = pq + ((long)z * NTOK + n) * MFEAT;
  const float* ks = ksum + (long)z * MFEAT;
  float den = 0.f;
  #pragma unroll 8
  for (int m = 0; m < MFEAT; ++m) den += (float)pr[m] * ks[m];
  const float inv = 1.f / den;
  const float* orow = obuf + ((long)z * NTOK + n) * DH;
  bf16* arow = attn + ((long)b * NTOK + n) * DDIM + h * DH;
  for (int e = 0; e < DH; ++e) arow[e] = (bf16)(orow[e] * inv);
}

__global__ __launch_bounds__(256) void mask_k(const float* __restrict__ x,
    const float* __restrict__ mw, const float* __restrict__ mb, float* __restrict__ out) {
  int idx = blockIdx.x * 256 + threadIdx.x;
  if (idx >= ROWS) return;
  int b = idx >> 9, n = idx & 511;
  const float* xr = x + (long)idx * DDIM;
  float a0 = 0, a1 = 0, a2 = 0, a3 = 0;
  for (int d = 0; d < DDIM; ++d) {
    float v = xr[d];
    a0 += v * mw[d * 4 + 0]; a1 += v * mw[d * 4 + 1];
    a2 += v * mw[d * 4 + 2]; a3 += v * mw[d * 4 + 3];
  }
  float lg[4] = { a0 + mb[0], a1 + mb[1], a2 + mb[2], a3 + mb[3] };
  #pragma unroll
  for (int j = 0; j < 4; ++j)
    out[((long)b * 4 + j) * NTOK + n] = 1.f / (1.f + expf(-lg[j]));
}

// weight (K,N) fp32 -> (N,K) bf16
__global__ __launch_bounds__(256) void wtrans_k(const float* __restrict__ w,
    bf16* __restrict__ wt, int K, int N) {
  long idx = (long)blockIdx.x * 256 + threadIdx.x;
  if (idx >= (long)K * N) return;
  int n = (int)(idx / K), k = (int)(idx % K);
  wt[idx] = (bf16)w[(long)k * N + n];
}

__global__ __launch_bounds__(256) void cvt_k(const float* __restrict__ src,
    bf16* __restrict__ dst, int cnt, float scale) {
  int i = blockIdx.x * 256 + threadIdx.x;
  if (i < cnt) dst[i] = (bf16)(src[i] * scale);
}

// ---------------- launch ----------------
extern "C" void kernel_launch(void* const* d_in, const int* in_sizes, int n_in,
                              void* d_out, int out_size, void* d_ws, size_t ws_size,
                              hipStream_t stream) {
  const float* mel     = (const float*)d_in[0];
  const float* patch_w = (const float*)d_in[1];
  const float* patch_b = (const float*)d_in[2];
  const float* pos     = (const float*)d_in[3];
  const float* proj    = (const float*)d_in[4];
  const float* ln1_s   = (const float*)d_in[5];
  const float* ln1_b   = (const float*)d_in[6];
  const float* qkv_w   = (const float*)d_in[7];
  const float* qkv_b   = (const float*)d_in[8];
  const float* ao_w    = (const float*)d_in[9];
  const float* ao_b    = (const float*)d_in[10];
  const float* ln2_s   = (const float*)d_in[11];
  const float* ln2_b   = (const float*)d_in[12];
  const float* ff1_w   = (const float*)d_in[13];
  const float* ff1_b   = (const float*)d_in[14];
  const float* ff2_w   = (const float*)d_in[15];
  const float* ff2_b   = (const float*)d_in[16];
  const float* mask_w  = (const float*)d_in[17];
  const float* mask_b  = (const float*)d_in[18];
  float* out = (float*)d_out;

  char* wsp = (char*)d_ws;
  auto alloc = [&](size_t bytes) -> char* {
    char* p = wsp; wsp += (bytes + 255) & ~(size_t)255; return p;
  };
  bf16*  wqkvT = (bf16*)alloc((size_t)6 * QKVW * DDIM * 2);
  bf16*  waoT  = (bf16*)alloc((size_t)6 * DDIM * DDIM * 2);
  bf16*  wff1T = (bf16*)alloc((size_t)6 * FFDIM * DDIM * 2);
  bf16*  wff2T = (bf16*)alloc((size_t)6 * DDIM * FFDIM * 2);
  bf16*  projb = (bf16*)alloc((size_t)MFEAT * DH * 2);
  float* x     = (float*)alloc((size_t)ROWS * DDIM * 4);
  bf16*  h     = (bf16*)alloc((size_t)ROWS * DDIM * 2);
  bf16*  qkv   = (bf16*)alloc((size_t)ROWS * QKVW * 2);
  float* u     = (float*)alloc((size_t)64 * NTOK * MFEAT * 4);
  bf16*  pq    = (bf16*)alloc((size_t)64 * NTOK * MFEAT * 2);
  bf16*  pkT   = (bf16*)alloc((size_t)64 * MFEAT * NTOK * 2);
  float* diagk = (float*)alloc((size_t)64 * NTOK * 4);
  float* ksum  = (float*)alloc((size_t)64 * MFEAT * 4);
  float* ctx   = (float*)alloc((size_t)64 * MFEAT * DH * 4);
  float* obuf  = (float*)alloc((size_t)64 * NTOK * DH * 4);
  bf16*  gbuf  = (bf16*)alloc((size_t)ROWS * FFDIM * 2);
  float* gmax  = (float*)alloc(256);

  dim3 blk(256);
  const float qscale = 1.0f / powf(96.0f, 0.25f);   // d^-1/4, folded into proj
  // weights -> bf16, transposed to (N,K)
  cvt_k<<<(MFEAT * DH + 255) / 256, blk, 0, stream>>>(proj, projb, MFEAT * DH, qscale);
  for (int l = 0; l < 6; ++l) {
    wtrans_k<<<(DDIM * QKVW + 255) / 256, blk, 0, stream>>>(qkv_w + (long)l * DDIM * QKVW,  wqkvT + (long)l * QKVW * DDIM, DDIM, QKVW);
    wtrans_k<<<(DDIM * DDIM + 255) / 256, blk, 0, stream>>>(ao_w  + (long)l * DDIM * DDIM,  waoT  + (long)l * DDIM * DDIM, DDIM, DDIM);
    wtrans_k<<<(DDIM * FFDIM + 255) / 256, blk, 0, stream>>>(ff1_w + (long)l * DDIM * FFDIM, wff1T + (long)l * FFDIM * DDIM, DDIM, FFDIM);
    wtrans_k<<<(FFDIM * DDIM + 255) / 256, blk, 0, stream>>>(ff2_w + (long)l * FFDIM * DDIM, wff2T + (long)l * DDIM * FFDIM, FFDIM, DDIM);
  }
  patch_k<<<(BATCH * NTOK * DDIM + 255) / 256, blk, 0, stream>>>(mel, patch_w, patch_b, pos, x);

  for (int l = 0; l < 6; ++l) {
    ln_k<<<ROWS, blk, 0, stream>>>(x, ln1_s + l * DDIM, ln1_b + l * DDIM, h);
    // qkv = h @ Wqkv + b  -> bf16
    gemm_k<bf16, true, 1, false><<<dim3(QKVW / 128, ROWS / 128, 1), blk, 0, stream>>>(
        h, 0, 0, DDIM,
        wqkvT + (long)l * QKVW * DDIM, 0, 0, DDIM,
        (void*)qkv, 0, 0, QKVW,
        qkv_b + l * QKVW, ROWS, QKVW, DDIM, 1, nullptr);
    // u_q = q @ (s*proj)^T  (batched over b,h)
    gemm_k<bf16, true, 0, false><<<dim3(2, 4, 64), blk, 0, stream>>>(
        qkv, (long)NTOK * QKVW, DH, QKVW,
        projb, 0, 0, DH,
        (void*)u, (long)HEADS * NTOK * MFEAT, (long)NTOK * MFEAT, MFEAT,
        nullptr, NTOK, MFEAT, DH, HEADS, nullptr);
    phi_q_k<<<(64 * NTOK) / 256, blk, 0, stream>>>(u, qkv, pq, diagk);
    init_gmax<<<1, 1, 0, stream>>>(gmax);
    // u_k, tracking global max (FAVOR+ key stabilizer)
    gemm_k<bf16, true, 0, true><<<dim3(2, 4, 64), blk, 0, stream>>>(
        qkv + DDIM, (long)NTOK * QKVW, DH, QKVW,
        projb, 0, 0, DH,
        (void*)u, (long)HEADS * NTOK * MFEAT, (long)NTOK * MFEAT, MFEAT,
        nullptr, NTOK, MFEAT, DH, HEADS, gmax);
    phi_k_k<<<(64 * MFEAT) / 256, blk, 0, stream>>>(u, diagk, gmax, pkT, ksum);
    // ctx = pk^T @ v  (A = pkT bf16, B = v slice of qkv, (K,N) staging)
    gemm_k<bf16, false, 0, false><<<dim3(1, 2, 64), blk, 0, stream>>>(
        pkT, (long)HEADS * MFEAT * NTOK, (long)MFEAT * NTOK, NTOK,
        qkv + 2 * DDIM, (long)NTOK * QKVW, DH, QKVW,
        (void*)ctx, (long)HEADS * MFEAT * DH, (long)MFEAT * DH, DH,
        nullptr, MFEAT, DH, NTOK, HEADS, nullptr);
    // out = pq @ ctx   (B = ctx fp32, (K,N) staging)
    gemm_k<float, false, 0, false><<<dim3(1, 4, 64), blk, 0, stream>>>(
        pq, (long)HEADS * NTOK * MFEAT, (long)NTOK * MFEAT, MFEAT,
        ctx, (long)HEADS * MFEAT * DH, (long)MFEAT * DH, DH,
        (void*)obuf, (long)HEADS * NTOK * DH, (long)NTOK * DH, DH,
        nullptr, NTOK, DH, MFEAT, HEADS, nullptr);
    merge_k<<<(64 * NTOK) / 256, blk, 0, stream>>>(obuf, pq, ksum, h);
    // x += attn @ Wao + b
    gemm_k<bf16, true, 2, false><<<dim3(DDIM / 128, ROWS / 128, 1), blk, 0, stream>>>(
        h, 0, 0, DDIM,
        waoT + (long)l * DDIM * DDIM, 0, 0, DDIM,
        (void*)x, 0, 0, DDIM,
        ao_b + l * DDIM, ROWS, DDIM, DDIM, 1, nullptr);
    ln_k<<<ROWS, blk, 0, stream>>>(x, ln2_s + l * DDIM, ln2_b + l * DDIM, h);
    // g = gelu(h @ Wff1 + b) -> bf16
    gemm_k<bf16, true, 3, false><<<dim3(FFDIM / 128, ROWS / 128, 1), blk, 0, stream>>>(
        h, 0, 0, DDIM,
        wff1T + (long)l * FFDIM * DDIM, 0, 0, DDIM,
        (void*)gbuf, 0, 0, FFDIM,
        ff1_b + l * FFDIM, ROWS, FFDIM, DDIM, 1, nullptr);
    // x += g @ Wff2 + b
    gemm_k<bf16, true, 2, false><<<dim3(DDIM / 128, ROWS / 128, 1), blk, 0, stream>>>(
        gbuf, 0, 0, FFDIM,
        wff2T + (long)l * DDIM * FFDIM, 0, 0, FFDIM,
        (void*)x, 0, 0, DDIM,
        ff2_b + l * DDIM, ROWS, DDIM, FFDIM, 1, nullptr);
  }
  mask_k<<<(ROWS + 255) / 256, blk, 0, stream>>>(x, mask_w, mask_b, out);
  (void)in_sizes; (void)n_in; (void)out_size; (void)ws_size;
}
